// RvuAttn_21784074125902
// MI455X (gfx1250) — matrix operand order repaired and checked by reading, not compiled, not run on hardware
//
#include <hip/hip_runtime.h>
#include <hip/hip_bf16.h>

// ---------------------------------------------------------------------------
// Fused scanline attention for MI455X (gfx1250, wave32, WMMA bf16).
// One workgroup (256 thr = 8 waves) per (b,h) scanline; X,Q,K,Vt,P staged in
// ~272 KB dynamic LDS. Weights pre-converted to bf16 and pre-swizzled into
// WMMA B-fragment order in d_ws by a prologue kernel, so the projection inner
// loop is pure {global_load_b128 x2, ds-resident A frags, v_wmma}.
// ---------------------------------------------------------------------------

#define Bn 8
#define Cn 256
#define Hn 128
#define Wn 128

typedef __attribute__((ext_vector_type(16))) __bf16 bf16x16;
typedef __attribute__((ext_vector_type(8)))  float  f32x8;

struct __attribute__((aligned(16))) U4 { unsigned x, y, z, w; };
struct __attribute__((aligned(16))) F4 { float x, y, z, w; };

union FragU {
    U4 q[2];
    unsigned short s[16];
    bf16x16 v;
};

// Native convert: v_cvt_pk_bf16_f32 class ops.
static __device__ __forceinline__ unsigned short f2bf(float f) {
    return __builtin_bit_cast(unsigned short, (__bf16)f);
}

// Load a 16x32 bf16 fragment (A layout; B layout mirrors it) from an LDS
// row-major matrix. All addresses are 16B aligned (pitch % 8 == 0, k0 % 8 == 0).
// lane<16 : row = rowBase+lane,    K-chunks {k0..k0+7,  k0+16..k0+23}
// lane>=16: row = rowBase+lane-16, K-chunks {k0+8..k0+15, k0+24..k0+31}
static __device__ __forceinline__ bf16x16
lds_frag(const unsigned short* base, int pitch, int rowBase, int k0, int lane) {
    int r = rowBase + (lane & 15);
    int c = k0 + ((lane >> 4) << 3);
    const U4* p = (const U4*)(base + r * pitch + c);   // 16B aligned
    FragU f;
    f.q[0] = p[0];        // ds_load_b128
    f.q[1] = p[2];        // ds_load_b128
    return f.v;
}

// Pre-swizzled weight fragment: contiguous 32 B per lane, coalesced 1 KB/wave.
static __device__ __forceinline__ bf16x16
wfrag_pre(const unsigned short* wsw, int p, int n, int ks, int lane) {
    const U4* src = (const U4*)(wsw + ((size_t)(((p * 16 + n) * 8 + ks) * 32 + lane)) * 16);
    FragU f;
    f.q[0] = src[0];
    f.q[1] = src[1];
    return f.v;
}

// LDS geometry (units: unsigned short). Pitches keep 16-lane b128 reads on
// distinct banks (264*2B = 132 dwords -> stride 4 banks mod 64).
#define PITCH_C 264          // rows with 256 bf16 columns
#define PITCH_W 136          // rows with 128 bf16 columns
#define XT_OFF  0
#define Q_OFF   (128 * PITCH_C)
#define K_OFF   (2 * 128 * PITCH_C)
#define VT_OFF  (3 * 128 * PITCH_C)
#define LDS_USHORTS (3 * 128 * PITCH_C + 256 * PITCH_W)
#define LDS_BYTES   (LDS_USHORTS * 2)   // 272,384 B < 320 KB

// ---------------------------------------------------------------------------
// Prologue: convert wq/wk/wv (f32, [o][c]) to bf16 in B-fragment order.
// ---------------------------------------------------------------------------
__global__ __launch_bounds__(256)
void swizzle_weights_kernel(const float* __restrict__ wq,
                            const float* __restrict__ wk,
                            const float* __restrict__ wv,
                            unsigned short* __restrict__ wsw) {
    int t = blockIdx.x * 256 + threadIdx.x;      // 0 .. 12287
    if (t >= 3 * 16 * 8 * 32) return;
    int lane = t & 31;
    int f    = t >> 5;
    int ks   = f & 7;
    int n    = (f >> 3) & 15;
    int p    = f >> 7;
    const float* wm = (p == 0) ? wq : ((p == 1) ? wk : wv);
    int o = 16 * n + (lane & 15);
    int c = 32 * ks + ((lane >> 4) << 3);
    const float* src = wm + o * Cn + c;
    FragU fr;
#pragma unroll
    for (int i = 0; i < 8; ++i) {
        fr.s[i]     = f2bf(src[i]);
        fr.s[i + 8] = f2bf(src[i + 16]);
    }
    U4* dst = (U4*)(wsw + (size_t)t * 16);
    dst[0] = fr.q[0];
    dst[1] = fr.q[1];
}

// ---------------------------------------------------------------------------
// Main fused kernel.
// ---------------------------------------------------------------------------
__global__ __launch_bounds__(256)
void RvuAttn_scanline_kernel(const float* __restrict__ a,
                             const float* __restrict__ bq,
                             const float* __restrict__ bk,
                             const float* __restrict__ bv,
                             const float* __restrict__ Wp,
                             const unsigned short* __restrict__ wsw,
                             float* __restrict__ out) {
    extern __shared__ unsigned short sm[];
    unsigned short* Xt = sm + XT_OFF;   // [128 w][256 c]   A for QKV GEMMs
    unsigned short* Qs = sm + Q_OFF;    // [128 w][256 o]   A for scores
    unsigned short* Ks = sm + K_OFF;    // [128 w][256 o]   B for scores
    unsigned short* Vt = sm + VT_OFF;   // [256 o][128 w]   B for P*V
    unsigned short* Pb = sm + XT_OFF;   // [128 q][128 k]   aliases dead Xt

    const int bh   = blockIdx.x;        // 0..1023
    const int bb   = bh >> 7;
    const int hh   = bh & (Hn - 1);
    const int tid  = threadIdx.x;
    const int lane = tid & 31;
    const int wave = tid >> 5;          // owns query rows 16*wave..16*wave+15
    const int ln15 = lane & 15;
    const int rHi  = (lane >> 4) << 3;  // +0 or +8 row offset of D fragments
    const int rowBase = 16 * wave + rHi;

    // ---- Phase 0: stage X slice (a[b,:,h,:]) transposed to Xt[w][c] bf16 ---
    {
        const int c = tid;              // one channel per thread
        const float* src = a + ((size_t)(bb * Cn + c) * Hn + hh) * Wn;
        unsigned short* dst = Xt + c;
#pragma unroll 4
        for (int j = 0; j < 32; ++j) {
            F4 v = *(const F4*)(src + 4 * j);
            dst[(4 * j + 0) * PITCH_C] = f2bf(v.x);
            dst[(4 * j + 1) * PITCH_C] = f2bf(v.y);
            dst[(4 * j + 2) * PITCH_C] = f2bf(v.z);
            dst[(4 * j + 3) * PITCH_C] = f2bf(v.w);
        }
    }
    __syncthreads();

    // ---- Phase 1: Q/K/V projections (wave computes its 16-row M tile) ------
    bf16x16 xa[8];
#pragma unroll
    for (int ks = 0; ks < 8; ++ks)
        xa[ks] = lds_frag(Xt, PITCH_C, 16 * wave, 32 * ks, lane);

    const float* Bi[3] = { bq, bk, bv };
#pragma unroll
    for (int p = 0; p < 3; ++p) {
        for (int n = 0; n < 16; ++n) {
            f32x8 acc = {};
#pragma unroll
            for (int ks = 0; ks < 8; ++ks) {
                bf16x16 bf = wfrag_pre(wsw, p, n, ks, lane);
                acc = __builtin_amdgcn_wmma_f32_16x16x32_bf16(
                        false, xa[ks], false, bf, (short)0, acc, false, false);
            }
            const float bias = Bi[p][16 * n + ln15];
            const int colO = 16 * n + ln15;
            if (p < 2) {
                // Row-major store, stride = PITCH_C -> immediate DS offsets.
                unsigned short* dp = (p == 0 ? Qs : Ks) + rowBase * PITCH_C + colO;
#pragma unroll
                for (int e = 0; e < 8; ++e)
                    dp[e * PITCH_C] = f2bf(acc[e] + bias);
            } else {
                // Transposed store: 8 consecutive bf16 -> one ds_store_b128.
                FragU pk;
#pragma unroll
                for (int e = 0; e < 8; ++e)
                    pk.s[e] = f2bf(acc[e] + bias);
                *(U4*)(Vt + colO * PITCH_W + rowBase) = pk.q[0];
            }
        }
    }
    __syncthreads();

    // ---- Phase 2: scores S = Q K^T * scale, softmax over keys --------------
    bf16x16 qa[8];
#pragma unroll
    for (int ks = 0; ks < 8; ++ks)
        qa[ks] = lds_frag(Qs, PITCH_C, 16 * wave, 32 * ks, lane);

    f32x8 sc[8];
#pragma unroll
    for (int kn = 0; kn < 8; ++kn) {
        f32x8 acc = {};
#pragma unroll
        for (int ks = 0; ks < 8; ++ks) {
            bf16x16 kf = lds_frag(Ks, PITCH_C, 16 * kn, 32 * ks, lane);
            acc = __builtin_amdgcn_wmma_f32_16x16x32_bf16(
                    false, qa[ks], false, kf, (short)0, acc, false, false);
        }
        sc[kn] = acc;
    }

    const float scale = 0.0625f;        // 1/sqrt(256)
#pragma unroll
    for (int e = 0; e < 8; ++e) {
        float mx = -3.0e38f;
#pragma unroll
        for (int kn = 0; kn < 8; ++kn) mx = fmaxf(mx, sc[kn][e]);
#pragma unroll
        for (int off = 8; off; off >>= 1) mx = fmaxf(mx, __shfl_xor(mx, off, 32));
        float sum = 0.0f;
#pragma unroll
        for (int kn = 0; kn < 8; ++kn) {
            float pv = __expf((sc[kn][e] - mx) * scale);
            sc[kn][e] = pv;
            sum += pv;
        }
#pragma unroll
        for (int off = 8; off; off >>= 1) sum += __shfl_xor(sum, off, 32);
        const float inv = 1.0f / sum;
        unsigned short* pp = Pb + (rowBase + e) * PITCH_W + ln15;
#pragma unroll
        for (int kn = 0; kn < 8; ++kn)
            pp[kn * 16] = f2bf(sc[kn][e] * inv);   // immediate DS offsets
    }
    // Pb rows are produced and consumed by the same wave; compiler-inserted
    // s_wait_dscnt covers the RAW, no workgroup barrier required.

    // ---- Phase 3: O = P V, fused residual epilogue -------------------------
    bf16x16 pa[4];
#pragma unroll
    for (int kk = 0; kk < 4; ++kk)
        pa[kk] = lds_frag(Pb, PITCH_W, 16 * wave, 32 * kk, lane);

    const float wp = Wp[0];
    for (int n = 0; n < 16; ++n) {
        f32x8 acc = {};
#pragma unroll
        for (int kk = 0; kk < 4; ++kk) {
            bf16x16 vf = lds_frag(Vt, PITCH_W, 16 * n, 32 * kk, lane);
            acc = __builtin_amdgcn_wmma_f32_16x16x32_bf16(
                    false, pa[kk], false, vf, (short)0, acc, false, false);
        }
        const int och = 16 * n + ln15;
        const size_t g = ((size_t)(bb * Cn + och) * Hn + hh) * Wn + rowBase;
        F4 a0 = *(const F4*)(a + g);
        F4 a1 = *(const F4*)(a + g + 4);
        F4 o0, o1;
        o0.x = a0.x + acc[0] * wp;  o0.y = a0.y + acc[1] * wp;
        o0.z = a0.z + acc[2] * wp;  o0.w = a0.w + acc[3] * wp;
        o1.x = a1.x + acc[4] * wp;  o1.y = a1.y + acc[5] * wp;
        o1.z = a1.z + acc[6] * wp;  o1.w = a1.w + acc[7] * wp;
        *(F4*)(out + g)     = o0;
        *(F4*)(out + g + 4) = o1;
    }
}

extern "C" void kernel_launch(void* const* d_in, const int* in_sizes, int n_in,
                              void* d_out, int out_size, void* d_ws, size_t ws_size,
                              hipStream_t stream) {
    (void)in_sizes; (void)n_in; (void)out_size; (void)ws_size;
    const float* a  = (const float*)d_in[0];
    const float* wq = (const float*)d_in[1];
    const float* bq = (const float*)d_in[2];
    const float* wk = (const float*)d_in[3];
    const float* bk = (const float*)d_in[4];
    const float* wv = (const float*)d_in[5];
    const float* bv = (const float*)d_in[6];
    const float* Wp = (const float*)d_in[7];
    float* out = (float*)d_out;
    unsigned short* wsw = (unsigned short*)d_ws;   // 384 KB of bf16 fragments

    // Prologue: 3*16*8*32 = 12288 threads.
    swizzle_weights_kernel<<<dim3(48), dim3(256), 0, stream>>>(wq, wk, wv, wsw);

    dim3 grid(Bn * Hn);   // 1024 scanlines
    dim3 block(256);      // 8 wave32
    RvuAttn_scanline_kernel<<<grid, block, LDS_BYTES, stream>>>(
        a, bq, bk, bv, Wp, wsw, out);
}
